// EncodeProcessDecode_25451976196335
// MI455X (gfx1250) — matrix operand order repaired and checked by reading, not compiled, not run on hardware
//
#include <hip/hip_runtime.h>

// ---------------------------------------------------------------------------
// EncodeProcessDecode GNN for gfx1250 (MI455X, wave32, WMMA).
//
// Each reference MLP = 3 stacked *linear* layers (no activation) + optional
// LayerNorm  =>  collapses exactly to one GEMM:
//   W_eff = W0@W1@W2 (f16, K zero-padded to mult. of 32, 128 output cols
//           -- decoder's 3 cols zero-padded to 128 so control flow around
//           WMMA is always uniform / EXEC all-ones)
//   b_eff = (b0@W1 + b1)@W2 + b2 (f32, zero-padded)
// Collapse recomputed on-device each call (deterministic, ~425 MFLOP).
//
// Fused kernel: 64-row x 128-col tile per 256-thread block (8 wave32s).
// Wave w -> row-tile (w>>1), col-half (w&1), 4 accumulators of
// v_wmma_f32_16x16x32_f16. K staged in 128-wide chunks:
//   x-tile: f16 LDS, stride 144 halves  -> A frag = 2x ds_load_b128
//   W-tile: f16 LDS pre-swizzled to B-fragment order -> B frag = 2x ds_load_b128
// Fused post-ops: bias, LayerNorm, residual add, segment-sum (f32 atomics).
//
// Param leaf order assumption (insertion-order flatten):
//   0 node_f, 1 edge_f, 2 senders, 3 receivers,
//   4..  enc_node{Ws0,Ws1,Ws2,bs0,bs1,bs2,g,b_ln}, 12.. enc_edge{..},
//   20+16i.. block_i{edge MLP(8), node MLP(8)}, 180.. dec{Ws0..2,bs0..2}
// ---------------------------------------------------------------------------

typedef __attribute__((ext_vector_type(16))) _Float16 v16h;
typedef __attribute__((ext_vector_type(8)))  _Float16 v8h;
typedef __attribute__((ext_vector_type(4)))  _Float16 v4h;
typedef __attribute__((ext_vector_type(8)))  float    v8f;
typedef __attribute__((ext_vector_type(4)))  float    v4f;

#define ROWS 64
#define XS   144                       // x LDS stride in halves (288B, 16B-aligned, 2-way-min banks)
#define XBYTES (ROWS * XS * 2)         // 18432
#define WOFF   XBYTES
#define WBYTES (4 * 8 * 2 * 256 * 2)   // 4 k32-chunks x 8 col-tiles x {lo,hi} x 32lanes x 8 halves
#define GEMM_BYTES (XBYTES + WBYTES)   // 51200 ; f32 y-tile (64*133*4=34048) aliases this
#define YS   133
#define STATS_OFF  GEMM_BYTES
#define SMEM_TOTAL (GEMM_BYTES + 2 * ROWS * 4)   // 51712 < 64KB

#define MODE_DIRECT  0
#define MODE_EDGECAT 1
#define MODE_NODECAT 2
#define POST_STORE 0
#define POST_ADD   1
#define POST_EDGE  2

__device__ __forceinline__ v16h cat8(v8h lo, v8h hi) {
    return __builtin_shufflevector(lo, hi, 0, 1, 2, 3, 4, 5, 6, 7,
                                   8, 9, 10, 11, 12, 13, 14, 15);
}

// ---------------- weight/bias collapse (f32, tiny) -------------------------

__global__ void collapse_w(const float* __restrict__ W0, const float* __restrict__ W1,
                           const float* __restrict__ W2, _Float16* __restrict__ Weff,
                           int K, int OC, int OCpad) {
    __shared__ float t1[128];
    const int r = blockIdx.x, t = threadIdx.x;
    float a = 0.f;
    if (r < K)
        for (int i = 0; i < 128; ++i) a += W0[r * 128 + i] * W1[i * 128 + t];
    t1[t] = a;
    __syncthreads();
    if (t < OCpad) {
        float o = 0.f;
        if (r < K && t < OC)
            for (int i = 0; i < 128; ++i) o += t1[i] * W2[i * OC + t];
        Weff[(long)r * OCpad + t] = (_Float16)o;
    }
}

__global__ void collapse_b(const float* __restrict__ b0, const float* __restrict__ W1,
                           const float* __restrict__ b1, const float* __restrict__ W2,
                           const float* __restrict__ b2, float* __restrict__ beff,
                           int OC, int OCpad) {
    __shared__ float tb[128];
    const int t = threadIdx.x;
    float a = 0.f;
    for (int i = 0; i < 128; ++i) a += b0[i] * W1[i * 128 + t];
    tb[t] = a + b1[t];
    __syncthreads();
    if (t < OCpad) {
        float o = 0.f;
        if (t < OC) {
            for (int i = 0; i < 128; ++i) o += tb[i] * W2[i * OC + t];
            o += b2[t];
        }
        beff[t] = o;
    }
}

// ---------------- fused collapsed-MLP kernel (WMMA) -------------------------

__global__ __launch_bounds__(256)
void mlp_fused(const float* __restrict__ x0, int in_dim,
               const float* __restrict__ nodes, const float* __restrict__ edges,
               const float* __restrict__ agg,
               const int* __restrict__ senders, const int* __restrict__ receivers,
               const _Float16* __restrict__ W, const float* __restrict__ bias,
               const float* __restrict__ ln_g, const float* __restrict__ ln_b,
               float* __restrict__ out, float* __restrict__ agg_out,
               int R, int Kpad, int out_cols,
               int mode, int post) {
    extern __shared__ char smem[];
    _Float16* xs   = (_Float16*)smem;
    _Float16* wsw  = (_Float16*)(smem + WOFF);     // swizzled B fragments
    float*    ysm  = (float*)smem;                 // aliases xs/wsw after GEMM
    float*    s_mu = (float*)(smem + STATS_OFF);
    float*    s_ri = s_mu + ROWS;

    const int t    = threadIdx.x;
    const int wave = t >> 5, lane = t & 31;
    const int rt = wave >> 1, ch = wave & 1;       // row tile, column half
    const int n16 = lane & 15, g = lane >> 4;
    const long rowbase = (long)blockIdx.x * ROWS;

    v8f acc[4];
    {
        v8f z = {};
        for (int j = 0; j < 4; ++j) acc[j] = z;
    }

    for (int kb = 0; kb < Kpad; kb += 128) {
        int kc = Kpad - kb; if (kc > 128) kc = 128;

        // ---- stage x chunk (f32 -> f16) ----
        if (mode == MODE_DIRECT && in_dim < 128) {
            // encoders: kc == 32, scalar with zero padding
            for (int i = t; i < ROWS * 32; i += 256) {
                int r = i >> 5, c = i & 31;
                long row = rowbase + r;
                float v = (row < R && c < in_dim) ? x0[row * in_dim + c] : 0.f;
                xs[r * XS + c] = (_Float16)v;
            }
        } else {
            // 128-wide source rows: vectorized float4 -> half4
            for (int i = t; i < ROWS * 32; i += 256) {
                int r = i >> 5, c4 = (i & 31) << 2;
                long row = rowbase + r;
                v4f f = {};
                if (row < R) {
                    const float* src;
                    if (mode == MODE_EDGECAT) {
                        int s = kb >> 7;   // 0: sender nodes, 1: receiver nodes, 2: edges
                        src = (s == 0) ? nodes + (long)senders[row] * 128
                            : (s == 1) ? nodes + (long)receivers[row] * 128
                                       : edges + row * 128;
                    } else if (mode == MODE_NODECAT) {
                        src = (kb == 0) ? nodes + row * 128 : agg + row * 128;
                    } else {
                        src = x0 + row * 128;
                    }
                    f = *(const v4f*)(src + c4);
                }
                v4h h4;
                #pragma unroll
                for (int j = 0; j < 4; ++j) h4[j] = (_Float16)f[j];
                *(v4h*)(xs + r * XS + c4) = h4;
            }
        }
        // ---- stage W chunk, swizzled into B-fragment order ----
        // region(kb32, ct, hsel): 32 lanes x 8 contiguous halves
        for (int i = t; i < kc * 16; i += 256) {
            int k  = i >> 4;                 // k within chunk
            int c8 = (i & 15) << 3;          // col: 0,8,...,120
            v8h w8 = *(const v8h*)(W + (size_t)(kb + k) * 128 + c8);
            int ct = c8 >> 4, n0 = c8 & 15;
            int gk = (k >> 3) & 1, h = (k & 7) | ((k >> 1) & 8);
            int base = ((((k >> 5) * 8 + ct) * 2 + (h >> 3)) << 8)
                     + ((gk * 16 + n0) << 3) + (h & 7);
            #pragma unroll
            for (int j = 0; j < 8; ++j) wsw[base + j * 8] = w8[j];
        }
        __syncthreads();

        // ---- WMMA accumulate over this K chunk (uniform control flow) ----
        const int rowl = rt * 16 + n16;
        for (int c32 = 0; c32 < kc; c32 += 32) {
            // A fragment: lane's halves = k in [8g,8g+7] U [16+8g,23+8g]
            const _Float16* ap = xs + rowl * XS + c32 + 8 * g;
            v16h a = cat8(*(const v8h*)ap, *(const v8h*)(ap + 16));
            const int kb32 = c32 >> 5;
            #pragma unroll
            for (int j = 0; j < 4; ++j) {
                int ct = ch * 4 + j;
                const _Float16* bp = wsw + (((kb32 * 8 + ct) * 2) << 8) + lane * 8;
                v16h b = cat8(*(const v8h*)bp, *(const v8h*)(bp + 256));
                acc[j] = __builtin_amdgcn_wmma_f32_16x16x32_f16(
                    false, a, false, b, (short)0, acc[j], false, false);
            }
        }
        __syncthreads();
    }

    // ---- accumulators (+bias) -> f32 LDS tile (aliased over xs/wsw) ----
    #pragma unroll
    for (int j = 0; j < 4; ++j) {
        int col = (ch * 4 + j) * 16 + n16;
        float bv = bias[col];
        #pragma unroll
        for (int v = 0; v < 8; ++v) {
            int rowl = rt * 16 + v + 8 * g;          // C/D layout: VGPR v -> M=v+8g
            ysm[rowl * YS + col] = acc[j][v] + bv;
        }
    }
    __syncthreads();

    // ---- LayerNorm stats (one thread per row) ----
    if (ln_g) {
        if (t < ROWS) {
            float s = 0.f, s2 = 0.f;
            for (int c = 0; c < 128; ++c) {
                float v = ysm[t * YS + c];
                s += v; s2 += v * v;
            }
            float mu  = s * (1.f / 128.f);
            float var = s2 * (1.f / 128.f) - mu * mu;
            s_mu[t] = mu;
            s_ri[t] = rsqrtf(var + 1e-5f);
        }
        __syncthreads();
    }

    // ---- apply LN + residual / store / segment-sum (coalesced) ----
    if (out_cols == 128) {
        for (int i = t; i < ROWS * 128; i += 256) {
            int r = i >> 7, c = i & 127;
            long row = rowbase + r;
            if (row < R) {
                float v = ysm[r * YS + c];
                if (ln_g) v = (v - s_mu[r]) * s_ri[r] * ln_g[c] + ln_b[c];
                if (post == POST_STORE) {
                    out[row * 128 + c] = v;
                } else if (post == POST_ADD) {
                    out[row * 128 + c] += v;                  // nodes += new_n
                } else {                                      // POST_EDGE
                    out[row * 128 + c] += v;                  // edges += new_e
                    atomicAdd(&agg_out[(long)receivers[row] * 128 + c], v);
                }
            }
        }
    } else {
        // decoder: first out_cols columns only, no LN
        for (int i = t; i < ROWS * out_cols; i += 256) {
            int r = i / out_cols, c = i - r * out_cols;
            long row = rowbase + r;
            if (row < R) out[row * out_cols + c] = ysm[r * YS + c];
        }
    }
}

// ---------------------------------------------------------------------------

extern "C" void kernel_launch(void* const* d_in, const int* in_sizes, int n_in,
                              void* d_out, int out_size, void* d_ws, size_t ws_size,
                              hipStream_t stream) {
    if (n_in < 186) return;  // unexpected flattening

    const float* node_f    = (const float*)d_in[0];
    const float* edge_f    = (const float*)d_in[1];
    const int*   senders   = (const int*)d_in[2];
    const int*   receivers = (const int*)d_in[3];
    const int N = in_sizes[0] / 12;
    const int E = in_sizes[2];
    const int P = 4;  // first param leaf

    auto F = [&](int i) { return (const float*)d_in[i]; };

    // ---- carve workspace (~53.5 MB) ----
    char* wp = (char*)d_ws;
    auto carve = [&](size_t bytes) {
        char* p = wp;
        wp += (bytes + 255) & ~(size_t)255;
        return (void*)p;
    };
    float* nodes = (float*)carve((size_t)N * 128 * 4);
    float* edges = (float*)carve((size_t)E * 128 * 4);
    float* agg   = (float*)carve((size_t)N * 128 * 4);
    _Float16* Wen = (_Float16*)carve(32 * 128 * 2);  float* ben = (float*)carve(128 * 4);
    _Float16* Wee = (_Float16*)carve(32 * 128 * 2);  float* bee = (float*)carve(128 * 4);
    _Float16* Wbe[10]; float* bbe[10];
    _Float16* Wbn[10]; float* bbn[10];
    for (int i = 0; i < 10; ++i) {
        Wbe[i] = (_Float16*)carve(384 * 128 * 2); bbe[i] = (float*)carve(128 * 4);
        Wbn[i] = (_Float16*)carve(256 * 128 * 2); bbn[i] = (float*)carve(128 * 4);
    }
    _Float16* Wd = (_Float16*)carve(128 * 128 * 2);  float* bd = (float*)carve(128 * 4);
    if ((size_t)(wp - (char*)d_ws) > ws_size) return;

    // ---- collapse all 23 MLPs ----
    auto collapse = [&](int base, _Float16* W, float* b, int K, int Kpad, int OC, int OCpad) {
        collapse_w<<<Kpad, 128, 0, stream>>>(F(base + 0), F(base + 1), F(base + 2), W, K, OC, OCpad);
        collapse_b<<<1, 128, 0, stream>>>(F(base + 3), F(base + 1), F(base + 4),
                                          F(base + 2), F(base + 5), b, OC, OCpad);
    };
    collapse(P + 0, Wen, ben, 12, 32, 128, 128);
    collapse(P + 8, Wee, bee, 7, 32, 128, 128);
    for (int i = 0; i < 10; ++i) {
        int eb = P + 16 + 16 * i, nb = eb + 8;
        collapse(eb, Wbe[i], bbe[i], 384, 384, 128, 128);
        collapse(nb, Wbn[i], bbn[i], 256, 256, 128, 128);
    }
    const int db = P + 176;                 // decoder: Ws0..2, bs0..2 (no LN)
    collapse(db, Wd, bd, 128, 128, 3, 128); // 3 real cols zero-padded to 128

    const size_t SMEM = SMEM_TOTAL;
    const int gN = (N + ROWS - 1) / ROWS, gE = (E + ROWS - 1) / ROWS;

    // ---- encoders ----
    mlp_fused<<<gN, 256, SMEM, stream>>>(node_f, 12, nullptr, nullptr, nullptr,
        nullptr, nullptr, Wen, ben, F(P + 6), F(P + 7),
        nodes, nullptr, N, 32, 128, MODE_DIRECT, POST_STORE);
    mlp_fused<<<gE, 256, SMEM, stream>>>(edge_f, 7, nullptr, nullptr, nullptr,
        nullptr, nullptr, Wee, bee, F(P + 14), F(P + 15),
        edges, nullptr, E, 32, 128, MODE_DIRECT, POST_STORE);

    // ---- 10 message-passing blocks ----
    for (int i = 0; i < 10; ++i) {
        int eb = P + 16 + 16 * i, nb = eb + 8;
        hipMemsetAsync(agg, 0, (size_t)N * 128 * 4, stream);
        // edge: new_e = LN(cat@We+be); edges += new_e; agg[recv] += new_e
        mlp_fused<<<gE, 256, SMEM, stream>>>(nullptr, 0, nodes, edges, nullptr,
            senders, receivers, Wbe[i], bbe[i], F(eb + 6), F(eb + 7),
            edges, agg, E, 384, 128, MODE_EDGECAT, POST_EDGE);
        // node: nodes += LN([nodes|agg]@Wn+bn)
        mlp_fused<<<gN, 256, SMEM, stream>>>(nullptr, 0, nodes, nullptr, agg,
            nullptr, nullptr, Wbn[i], bbn[i], F(nb + 6), F(nb + 7),
            nodes, nullptr, N, 256, 128, MODE_NODECAT, POST_ADD);
    }

    // ---- decoder (no LN; 3 cols, weights padded to 128) ----
    mlp_fused<<<gN, 256, SMEM, stream>>>(nodes, 128, nullptr, nullptr, nullptr,
        nullptr, nullptr, Wd, bd, nullptr, nullptr,
        (float*)d_out, nullptr, N, 128, 3, MODE_DIRECT, POST_STORE);
}
